// STA_Module_31920196944375
// MI455X (gfx1250) — matrix operand order repaired and checked by reading, not compiled, not run on hardware
//
#include <hip/hip_runtime.h>

// Problem constants (reference: B=4, C=512, H=W=64)
#define B_   4
#define C_   512
#define N_   4096      // H*W
#define CQ_  64        // C/8
#define SROW 4100      // 4096 + 4 pad: stride%64banks = 4 -> halfwave rows (m, m+8)
                       // differ by 32 banks, conflict-free WMMA-tile stores

typedef __attribute__((ext_vector_type(16))) __bf16        v16bf;
typedef __attribute__((ext_vector_type(8)))  float         v8f;
typedef __attribute__((ext_vector_type(4)))  float         f32x4;
typedef __attribute__((ext_vector_type(4)))  unsigned int  u32x4;

// Fragment: 16 bf16 = 32B = 8 VGPRs. POD union (ext_vector members only).
union BFU {
    v16bf v;
    u32x4 q[2];        // two 16B chunks for vector loads
};

__device__ inline v8f wmma_bf16(const BFU& a, const BFU& b, v8f c) {
    return __builtin_amdgcn_wmma_f32_16x16x32_bf16(
        /*neg_a=*/false, a.v, /*neg_b=*/false, b.v,
        /*c_mod=*/(short)0, c, /*reuse_a=*/false, /*reuse_b=*/false);
}

// native float->bf16 (lowers to v_cvt_pk_bf16_f32 class ops on gfx1250)
__device__ inline void pack8(BFU& d, int base, f32x4 a, f32x4 b) {
    d.v[base + 0] = (__bf16)a.x; d.v[base + 1] = (__bf16)a.y;
    d.v[base + 2] = (__bf16)a.z; d.v[base + 3] = (__bf16)a.w;
    d.v[base + 4] = (__bf16)b.x; d.v[base + 5] = (__bf16)b.y;
    d.v[base + 6] = (__bf16)b.z; d.v[base + 7] = (__bf16)b.w;
}

// ---------------------------------------------------------------------------
// Projection Q/K: Y[b][n][o] (bf16, o in [0,64)) =
//   scale * (sum_c W[o][c] * X[b][c][n] + bias[o])
// scale = 0.125 for Q (softmax scale folded in), 1.0 for K.
// grid = B * N/16 blocks, 128 threads (4 waves, one 16x16 o-tile each).
// ---------------------------------------------------------------------------
__global__ __launch_bounds__(128)
void proj_qk_kernel(const float* __restrict__ X, const float* __restrict__ W,
                    const float* __restrict__ bias, __bf16* __restrict__ Y,
                    float scale) {
    const int b  = blockIdx.x >> 8;           // 256 n-tiles per batch
    const int n0 = (blockIdx.x & 255) << 4;
    const int tid = threadIdx.x;
    const int wv = tid >> 5, L = tid & 31, lh = L & 15, g = L >> 4;
    const int o0 = wv * 16;

    v8f acc = {0.f,0.f,0.f,0.f,0.f,0.f,0.f,0.f};
    for (int ks = 0; ks < 16; ++ks) {
        const int kbase = ks * 32;
        // A fragment: A[m][k] = X[kbase+k][n0+m], m = lh
        BFU a;
        #pragma unroll
        for (int e = 0; e < 16; ++e) {
            int c = kbase + g * 16 + e;
            a.v[e] = (__bf16)X[((size_t)(b * C_ + c)) * N_ + n0 + lh];
        }
        // B fragment: B[k][o] = W[o0+lh][kbase + 16g + e] -> 16 contiguous fp32
        const f32x4* wrow = (const f32x4*)(W + (size_t)(o0 + lh) * C_ + kbase + g * 16);
        BFU bf;
        pack8(bf, 0, wrow[0], wrow[1]);
        pack8(bf, 8, wrow[2], wrow[3]);
        acc = wmma_bf16(a, bf, acc);
    }
    const float bo = bias[o0 + lh];
    #pragma unroll
    for (int v = 0; v < 8; ++v) {
        int n = n0 + v + 8 * g;
        Y[((size_t)(b * N_) + n) * CQ_ + o0 + lh] = (__bf16)((acc[v] + bo) * scale);
    }
}

// ---------------------------------------------------------------------------
// Projection V (dual): V[b][c][n] (bf16, transposed layout) =
//   sum_c' Wvs[c][c']*Xs[b][c'][n] + Wvt[c][c']*Xt[b][c'][n] + bvs[c] + bvt[c]
// grid = B * N/16 blocks, 256 threads (8 waves x 4 o-tiles = 32 tiles = 512 ch).
// X tiles staged through LDS; output staged through LDS for contiguous stores.
// ---------------------------------------------------------------------------
__global__ __launch_bounds__(256)
void proj_v_kernel(const float* __restrict__ X1, const float* __restrict__ X2,
                   const float* __restrict__ W1, const float* __restrict__ W2,
                   const float* __restrict__ b1, const float* __restrict__ b2,
                   __bf16* __restrict__ V) {
    __shared__ float  xtile[32 * 17];          // [c_local][n_local], pad 17
    __shared__ __bf16 ot[512 * 16];            // [c][m] staging, 16 KB

    const int b  = blockIdx.x >> 8;
    const int n0 = (blockIdx.x & 255) << 4;
    const int tid = threadIdx.x;
    const int wv = tid >> 5, L = tid & 31, lh = L & 15, g = L >> 4;

    v8f acc[4];
    #pragma unroll
    for (int i = 0; i < 4; ++i) acc[i] = (v8f){0.f,0.f,0.f,0.f,0.f,0.f,0.f,0.f};

    for (int pass = 0; pass < 2; ++pass) {
        const float* X = pass ? X2 : X1;
        const float* W = pass ? W2 : W1;
        for (int ks = 0; ks < 16; ++ks) {
            const int kbase = ks * 32;
            __syncthreads();
            #pragma unroll
            for (int r = 0; r < 2; ++r) {              // stage 32c x 16n fp32 tile
                int idx = tid * 2 + r;                 // 0..511, coalesced over n
                int cl = idx >> 4, nl = idx & 15;
                xtile[cl * 17 + nl] = X[((size_t)(b * C_ + kbase + cl)) * N_ + n0 + nl];
            }
            __syncthreads();
            BFU a;                                     // A[m][k] from LDS
            #pragma unroll
            for (int e = 0; e < 16; ++e) {
                int k = g * 16 + e;
                a.v[e] = (__bf16)xtile[k * 17 + lh];
            }
            #pragma unroll
            for (int oti = 0; oti < 4; ++oti) {
                int o = (wv + oti * 8) * 16 + lh;      // output channel
                const f32x4* wrow = (const f32x4*)(W + (size_t)o * C_ + kbase + g * 16);
                BFU bf;
                pack8(bf, 0, wrow[0], wrow[1]);
                pack8(bf, 8, wrow[2], wrow[3]);
                acc[oti] = wmma_bf16(a, bf, acc[oti]);
            }
        }
    }
    // bias + stage to LDS [c][m]
    #pragma unroll
    for (int oti = 0; oti < 4; ++oti) {
        int c = (wv + oti * 8) * 16 + lh;
        float bo = b1[c] + b2[c];
        #pragma unroll
        for (int v = 0; v < 8; ++v)
            ot[c * 16 + v + 8 * g] = (__bf16)(acc[oti][v] + bo);
    }
    __syncthreads();
    // contiguous 32B stores per channel row: V[b][c][n0..n0+16)
    #pragma unroll
    for (int r = 0; r < 2; ++r) {
        int c = tid * 2 + r;
        const u32x4* src = (const u32x4*)(ot + c * 16);
        u32x4* dst = (u32x4*)(V + ((size_t)(b * C_ + c)) * N_ + n0);
        dst[0] = src[0];
        dst[1] = src[1];
    }
}

// ---------------------------------------------------------------------------
// Flash attention + residual epilogue.
// grid = B * N/16, 256 threads. Dynamic LDS: 16 x SROW fp32 score strip (~257 KB,
// fits MI455X's 320 KB WGP LDS). Pass1: S = Qs*K^T (softmax scale pre-folded into
// Q). Cooperative rowmax/rowsum (f32x4 LDS reads). Pass2: O = P*V, exp on the
// fly, each wave owns 64 value channels. Epilogue: 1/sum, +x_spatial+x_temporal,
// coalesced float4 stores via LDS transpose (overlays S).
// ---------------------------------------------------------------------------
__global__ __launch_bounds__(256)
void flash_kernel(const __bf16* __restrict__ Q,
                  const __bf16* __restrict__ K,
                  const __bf16* __restrict__ V,
                  const float* __restrict__ xs, const float* __restrict__ xt,
                  float* __restrict__ out) {
    extern __shared__ float S[];          // [16][SROW] fp32 scores
    __shared__ float red[256];
    __shared__ float rowmax[16];
    __shared__ float rowinv[16];

    const int b  = blockIdx.x >> 8;
    const int i0 = (blockIdx.x & 255) << 4;     // query row base
    const int tid = threadIdx.x;
    const int wv = tid >> 5, L = tid & 31, lh = L & 15, g = L >> 4;

    // ---- Q fragments (16 rows x 64 cq = 2 A-fragments), contiguous b128 loads
    BFU qa[2];
    {
        const u32x4* qrow = (const u32x4*)(Q + ((size_t)(b * N_) + i0 + lh) * CQ_);
        #pragma unroll
        for (int f = 0; f < 2; ++f) {
            qa[f].q[0] = qrow[f * 4 + g];          // cq = 32f + 8g + 0..7
            qa[f].q[1] = qrow[f * 4 + 2 + g];      // cq = 32f + 16 + 8g + 0..7
        }
    }

    // ---- Pass 1: score strip. Wave wv owns columns [wv*512, wv*512+512)
    {
        const int jb = wv * 512;
        for (int t = 0; t < 32; ++t) {
            const int j0 = jb + t * 16;
            const u32x4* krow = (const u32x4*)(K + ((size_t)(b * N_) + j0 + lh) * CQ_);
            BFU kb0, kb1;                           // B[k][n]: n=lh (key col), k=cq
            kb0.q[0] = krow[g];     kb0.q[1] = krow[2 + g];
            kb1.q[0] = krow[4 + g]; kb1.q[1] = krow[6 + g];
            v8f acc = {0.f,0.f,0.f,0.f,0.f,0.f,0.f,0.f};
            acc = wmma_bf16(qa[0], kb0, acc);
            acc = wmma_bf16(qa[1], kb1, acc);
            #pragma unroll
            for (int v = 0; v < 8; ++v)
                S[(v + 8 * g) * SROW + j0 + lh] = acc[v];
        }
    }
    __syncthreads();

    // ---- Softmax statistics: 16 threads per row, 256 cols each (f32x4 reads)
    {
        const int row = tid >> 4, ch = tid & 15;
        const f32x4* sp4 = (const f32x4*)(S + row * SROW + ch * 256);
        float mx = -3.4e38f;
        for (int c = 0; c < 64; ++c) {
            f32x4 t = sp4[c];
            mx = fmaxf(mx, fmaxf(fmaxf(t.x, t.y), fmaxf(t.z, t.w)));
        }
        red[tid] = mx;
        __syncthreads();
        if (tid < 16) {
            float m2 = red[tid * 16];
            for (int c = 1; c < 16; ++c) m2 = fmaxf(m2, red[tid * 16 + c]);
            rowmax[tid] = m2;
        }
        __syncthreads();
        const float m = rowmax[row];
        float sum = 0.f;
        for (int c = 0; c < 64; ++c) {
            f32x4 t = sp4[c];
            sum += __expf(t.x - m) + __expf(t.y - m) + __expf(t.z - m) + __expf(t.w - m);
        }
        red[tid] = sum;
        __syncthreads();
        if (tid < 16) {
            float s2 = 0.f;
            for (int c = 0; c < 16; ++c) s2 += red[tid * 16 + c];
            rowinv[tid] = 1.f / s2;
        }
        __syncthreads();
    }

    // ---- Pass 2: O = P * V. Wave wv owns channels [wv*64, wv*64+64)
    const float qmax = rowmax[lh];        // A-fragment rows are m = lh
    v8f acc[4];
    #pragma unroll
    for (int i = 0; i < 4; ++i) acc[i] = (v8f){0.f,0.f,0.f,0.f,0.f,0.f,0.f,0.f};
    const int c0 = wv * 64;
    for (int j0 = 0; j0 < N_; j0 += 32) {
        BFU pa;                            // P tile 16x32, exp on the fly
        #pragma unroll
        for (int e = 0; e < 16; ++e) {
            int jj = j0 + g * 16 + e;
            pa.v[e] = (__bf16)__expf(S[lh * SROW + jj] - qmax);
        }
        #pragma unroll
        for (int oti = 0; oti < 4; ++oti) {
            int c = c0 + oti * 16 + lh;    // B[k=j][n=c] from V[b][c][j]: contiguous j
            const u32x4* vrow = (const u32x4*)(V + ((size_t)(b * C_ + c)) * N_ + j0 + 16 * g);
            BFU vb;
            vb.q[0] = vrow[0];
            vb.q[1] = vrow[1];
            if (oti == 0)                  // stream-ahead hint (global_prefetch_b8)
                __builtin_prefetch((const void*)(vrow + 16), 0, 1);
            acc[oti] = wmma_bf16(pa, vb, acc[oti]);
        }
    }
    __syncthreads();                       // all waves done reading S

    // ---- Normalize + stage transpose (overlay S region), then coalesced store
    float* OT = S;                         // [c][m], 512*16 fp32 = 32 KB
    #pragma unroll
    for (int oti = 0; oti < 4; ++oti) {
        int c = c0 + oti * 16 + lh;
        #pragma unroll
        for (int v = 0; v < 8; ++v) {
            int m = v + 8 * g;
            OT[c * 16 + m] = acc[oti][v] * rowinv[m];
        }
    }
    __syncthreads();
    #pragma unroll
    for (int r = 0; r < 2; ++r) {
        int c = tid * 2 + r;
        size_t base = ((size_t)(b * C_ + c)) * N_ + i0;
        const f32x4* o4 = (const f32x4*)(OT + c * 16);
        const f32x4* a4 = (const f32x4*)(xs + base);
        const f32x4* b4 = (const f32x4*)(xt + base);
        f32x4* d4 = (f32x4*)(out + base);
        #pragma unroll
        for (int q = 0; q < 4; ++q)
            d4[q] = o4[q] + a4[q] + b4[q];   // + both residuals
    }
}

// ---------------------------------------------------------------------------
extern "C" void kernel_launch(void* const* d_in, const int* in_sizes, int n_in,
                              void* d_out, int out_size, void* d_ws, size_t ws_size,
                              hipStream_t stream) {
    (void)in_sizes; (void)n_in; (void)out_size; (void)ws_size;
    const float* xs  = (const float*)d_in[0];
    const float* xt  = (const float*)d_in[1];
    const float* Wq  = (const float*)d_in[2];
    const float* bq  = (const float*)d_in[3];
    const float* Wk  = (const float*)d_in[4];
    const float* bk  = (const float*)d_in[5];
    const float* Wvs = (const float*)d_in[6];
    const float* bvs = (const float*)d_in[7];
    const float* Wvt = (const float*)d_in[8];
    const float* bvt = (const float*)d_in[9];
    float* out = (float*)d_out;

    __bf16* Qb = (__bf16*)d_ws;                           // (B,N,64) bf16
    __bf16* Kb = Qb + (size_t)B_ * N_ * CQ_;              // (B,N,64) bf16
    __bf16* Vb = Kb + (size_t)B_ * N_ * CQ_;              // (B,C,N) bf16

    dim3 grid(B_ * (N_ / 16));
    proj_qk_kernel<<<grid, 128, 0, stream>>>(xs, Wq, bq, Qb, 0.125f); // fold CQ^-0.5
    proj_qk_kernel<<<grid, 128, 0, stream>>>(xt, Wk, bk, Kb, 1.0f);
    proj_v_kernel<<<grid, 256, 0, stream>>>(xs, xt, Wvs, Wvt, bvs, bvt, Vb);

    size_t sbytes = (size_t)16 * SROW * sizeof(float);    // ~256 KB LDS
    (void)hipFuncSetAttribute(reinterpret_cast<const void*>(flash_kernel),
                              hipFuncAttributeMaxDynamicSharedMemorySize, (int)sbytes);
    flash_kernel<<<grid, 256, sbytes, stream>>>(Qb, Kb, Vb, xs, xt, out);
}
